// SkipGramModel_6579889898198
// MI455X (gfx1250) — compile-verified, matrix operand order checked by reading, not run on hardware
//
#include <hip/hip_runtime.h>
#include <hip/hip_bf16.h>
#include <hip/hip_fp16.h>

typedef __attribute__((ext_vector_type(16))) _Float16 v16h;
typedef __attribute__((ext_vector_type(8)))  float    v8f;

#define EMBED_DIM 128
#define NEG_NUM   5
#define ELEMS_PER_WAVE 16
#define WAVES_PER_BLOCK 8
#define BLOCK_THREADS 256

__device__ __forceinline__ float log_sigmoid(float x) {
    // numerically stable: min(x,0) - log1p(exp(-|x|))
    return fminf(x, 0.0f) - log1pf(expf(-fabsf(x)));
}

// Load one K=32 chunk of a row in the 16-bit WMMA A/B register layout.
// Lane (m = lane&15, h = lane>>4) holds K = [h*8, h*8+8) in elems 0..7 and
// K = [16+h*8, 16+h*8+8) in elems 8..15 (CDNA5 ISA, 16-bit A 16x32 table).
__device__ __forceinline__ v16h load_chunk(const float* __restrict__ row,
                                           int kbase, int h) {
    const float4* p0 = reinterpret_cast<const float4*>(row + kbase + h * 8);
    const float4* p1 = reinterpret_cast<const float4*>(row + kbase + 16 + h * 8);
    float4 a = p0[0], b = p0[1];
    float4 c = p1[0], d = p1[1];
    v16h r;
    r[0]  = (_Float16)a.x; r[1]  = (_Float16)a.y; r[2]  = (_Float16)a.z; r[3]  = (_Float16)a.w;
    r[4]  = (_Float16)b.x; r[5]  = (_Float16)b.y; r[6]  = (_Float16)b.z; r[7]  = (_Float16)b.w;
    r[8]  = (_Float16)c.x; r[9]  = (_Float16)c.y; r[10] = (_Float16)c.z; r[11] = (_Float16)c.w;
    r[12] = (_Float16)d.x; r[13] = (_Float16)d.y; r[14] = (_Float16)d.z; r[15] = (_Float16)d.w;
    return r;
}

__global__ __launch_bounds__(BLOCK_THREADS) void skipgram_wmma_kernel(
    const float* __restrict__ emb,
    const int*   __restrict__ centers,
    const int*   __restrict__ contexts,
    const int*   __restrict__ negs,
    float*       __restrict__ partial) {
    const int tid  = threadIdx.x;
    const int lane = tid & 31;
    const int wave = tid >> 5;
    const int m    = lane & 15;   // row/col index within 16x16 tile
    const int h    = lane >> 4;   // half-select for K striping
    const int base = ((int)blockIdx.x * WAVES_PER_BLOCK + wave) * ELEMS_PER_WAVE;
    const int b    = base + m;    // batch element owned by this lane (dup for h=1)

    // Row base pointers (per lane): center row for A, context + 5 negative rows for B.
    const float* rowU = emb + (size_t)centers[b]  * EMBED_DIM;
    const float* rowV = emb + (size_t)contexts[b] * EMBED_DIM;
    const float* rowN0 = emb + (size_t)negs[b * NEG_NUM + 0] * EMBED_DIM;
    const float* rowN1 = emb + (size_t)negs[b * NEG_NUM + 1] * EMBED_DIM;
    const float* rowN2 = emb + (size_t)negs[b * NEG_NUM + 2] * EMBED_DIM;
    const float* rowN3 = emb + (size_t)negs[b * NEG_NUM + 3] * EMBED_DIM;
    const float* rowN4 = emb + (size_t)negs[b * NEG_NUM + 4] * EMBED_DIM;

    v8f accP = {};  // diag -> positive scores u.v
    v8f accN = {};  // diag -> sum_k u.neg_v[k]  (diag of sum == sum of diags)

#pragma unroll
    for (int c = 0; c < EMBED_DIM / 32; ++c) {
        const int kbase = c * 32;
        v16h A  = load_chunk(rowU,  kbase, h);
        v16h Bv = load_chunk(rowV,  kbase, h);
        accP = __builtin_amdgcn_wmma_f32_16x16x32_f16(false, A, false, Bv,
                                                      (short)0, accP, false, false);
        v16h B0 = load_chunk(rowN0, kbase, h);
        accN = __builtin_amdgcn_wmma_f32_16x16x32_f16(false, A, false, B0,
                                                      (short)0, accN, false, false);
        v16h B1 = load_chunk(rowN1, kbase, h);
        accN = __builtin_amdgcn_wmma_f32_16x16x32_f16(false, A, false, B1,
                                                      (short)0, accN, false, false);
        v16h B2 = load_chunk(rowN2, kbase, h);
        accN = __builtin_amdgcn_wmma_f32_16x16x32_f16(false, A, false, B2,
                                                      (short)0, accN, false, false);
        v16h B3 = load_chunk(rowN3, kbase, h);
        accN = __builtin_amdgcn_wmma_f32_16x16x32_f16(false, A, false, B3,
                                                      (short)0, accN, false, false);
        v16h B4 = load_chunk(rowN4, kbase, h);
        accN = __builtin_amdgcn_wmma_f32_16x16x32_f16(false, A, false, B4,
                                                      (short)0, accN, false, false);
    }

    // Extract diagonals: D[m][m] sits in VGPR (m&7) at lane (m<8 ? m : m+16).
    float pos = 0.0f, neg = 0.0f;
#pragma unroll
    for (int r = 0; r < 8; ++r) {
        float pa = __shfl(accP[r], r, 32);        // diag(r)
        float pb = __shfl(accP[r], r + 24, 32);   // diag(r+8): N=r+8 -> lane r+24
        float na = __shfl(accN[r], r, 32);
        float nb = __shfl(accN[r], r + 24, 32);
        if (lane == r)     { pos = pa; neg = na; }
        if (lane == r + 8) { pos = pb; neg = nb; }
    }

    float loss = 0.0f;
    if (lane < 16) {
        loss = log_sigmoid(pos) + log_sigmoid(-neg);
    }

    // Wave32 reduction
#pragma unroll
    for (int off = 16; off > 0; off >>= 1) {
        loss += __shfl_xor(loss, off, 32);
    }

    __shared__ float smem[WAVES_PER_BLOCK];
    if (lane == 0) smem[wave] = loss;
    __syncthreads();
    if (tid == 0) {
        float s = 0.0f;
#pragma unroll
        for (int i = 0; i < WAVES_PER_BLOCK; ++i) s += smem[i];
        partial[blockIdx.x] = s;
    }
}

// Deterministic fixed-order final reduction over the (tiny) partial array.
__global__ void skipgram_finalize_kernel(const float* __restrict__ partial,
                                         float* __restrict__ out, int n) {
    if (threadIdx.x == 0 && blockIdx.x == 0) {
        float s = 0.0f;
        for (int i = 0; i < n; ++i) s += partial[i];
        out[0] = -s;
    }
}

extern "C" void kernel_launch(void* const* d_in, const int* in_sizes, int n_in,
                              void* d_out, int out_size, void* d_ws, size_t ws_size,
                              hipStream_t stream) {
    const float* emb      = (const float*)d_in[0];
    const int*   centers  = (const int*)d_in[1];
    const int*   contexts = (const int*)d_in[2];
    const int*   negs     = (const int*)d_in[3];
    float*       out      = (float*)d_out;
    float*       partial  = (float*)d_ws;

    const int batch  = in_sizes[1];                               // 16384
    const int blocks = batch / (WAVES_PER_BLOCK * ELEMS_PER_WAVE); // 128

    skipgram_wmma_kernel<<<blocks, BLOCK_THREADS, 0, stream>>>(
        emb, centers, contexts, negs, partial);
    skipgram_finalize_kernel<<<1, 32, 0, stream>>>(partial, out, blocks);
}
